// GATGFTFSharedEncoder_75780402970755
// MI455X (gfx1250) — matrix operand order repaired and checked by reading, not compiled, not run on hardware
//
#include <hip/hip_runtime.h>
#include <hip/hip_bf16.h>
#include <math.h>

typedef __attribute__((ext_vector_type(16))) _Float16 v16h;
typedef __attribute__((ext_vector_type(8)))  _Float16 v8h;
typedef __attribute__((ext_vector_type(8)))  float    v8f;

static constexpr int kH   = 64;
static constexpr int kNP  = 32768;
static constexpr int kNT  = 131072;
static constexpr int kNF  = 131072;
static constexpr int kE   = 524288;
static constexpr int kB   = 512;

// ---------------------------------------------------------------- utilities
__global__ void k_fill(float* __restrict__ p, long n, float v) {
  long i = (long)blockIdx.x * blockDim.x + threadIdx.x;
  long stride = (long)gridDim.x * blockDim.x;
  for (; i < n; i += stride) p[i] = v;
}

__global__ void k_cvt_f16(const float* __restrict__ s, _Float16* __restrict__ d, long n) {
  long i = (long)blockIdx.x * blockDim.x + threadIdx.x;
  long stride = (long)gridDim.x * blockDim.x;
  for (; i < n; i += stride) d[i] = (_Float16)s[i];
}

// convert 64x64 weight matrices to f16, TRANSPOSED: dst[mat][n][k] = src[mat][k][n]
__global__ void k_cvt_wT(const float* __restrict__ s, _Float16* __restrict__ d, long nmat) {
  long i = (long)blockIdx.x * blockDim.x + threadIdx.x;
  long stride = (long)gridDim.x * blockDim.x;
  long tot = nmat << 12;
  for (; i < tot; i += stride) {
    long mat = i >> 12;
    int  n   = (int)((i >> 6) & 63);
    int  k   = (int)(i & 63);
    d[i] = (_Float16)s[(mat << 12) + (long)k * 64 + n];
  }
}

// ---------------------------------------------------------------- embedding
__global__ void k_embed(const float* __restrict__ mass, const int* __restrict__ part_x,
                        const float* __restrict__ gw, const float* __restrict__ stab,
                        float* __restrict__ out) {
  int i = blockIdx.x * blockDim.x + threadIdx.x;
  if (i >= kNP) return;
  float mv = mass[i];
  int s = part_x[i];
  float* o = out + (size_t)i * kH;
#pragma unroll
  for (int j = 0; j < 32; ++j) o[j] = mv * gw[j];
#pragma unroll
  for (int j = 0; j < 32; ++j) o[32 + j] = stab[s * 32 + j];
}

// ---------------------------------------------------------------- WMMA GEMM
// H[N,64] = f32( X16[N,64] @ W[64,64] ), W provided transposed (Wt[n][k]).
// 128 rows/block, 8 waves; per wave: 16-row tile, 4 col tiles, K=64 -> 8 WMMAs.
// X tile and Wt staged into LDS via CDNA5 async global->LDS DMA (ASYNCcnt).
__global__ __launch_bounds__(256)
void k_gemm64(const _Float16* __restrict__ X16, const _Float16* __restrict__ Wt,
              float* __restrict__ Hout, int nrows) {
  __shared__ alignas(16) _Float16 sX[128 * 64];
  __shared__ alignas(16) _Float16 sWt[64 * 64];
  int tid = threadIdx.x;
  long row0 = (long)blockIdx.x * 128;

  // LDS byte offsets = low 32 bits of flat shared address (ISA 10.2)
  unsigned ldsX = (unsigned)(unsigned long long)(&sX[0]);
  unsigned ldsW = (unsigned)(unsigned long long)(&sWt[0]);
  const _Float16* gX = X16 + row0 * kH;
#pragma unroll
  for (int c = 0; c < 4; ++c) {              // 128*64 halves = 1024 x 16B chunks
    int q = tid + 256 * c;
    unsigned long long ga = (unsigned long long)(gX + q * 8);
    unsigned la = ldsX + q * 16;
    asm volatile("global_load_async_to_lds_b128 %0, %1, off" :: "v"(la), "v"(ga) : "memory");
  }
#pragma unroll
  for (int c = 0; c < 2; ++c) {              // 64*64 halves = 512 x 16B chunks
    int q = tid + 256 * c;
    unsigned long long ga = (unsigned long long)(Wt + q * 8);
    unsigned la = ldsW + q * 16;
    asm volatile("global_load_async_to_lds_b128 %0, %1, off" :: "v"(la), "v"(ga) : "memory");
  }
  asm volatile("s_wait_asynccnt 0x0" ::: "memory");
  __syncthreads();

  int wave = tid >> 5;
  int lane = tid & 31;
  int hl   = lane >> 4;
  int m    = lane & 15;

  union V16 { v16h v; v8h h[2]; };

  // A fragment (ISA 16x32 f16 layout): contiguous 8-half runs per lane half
  const _Float16* xrow = &sX[(wave * 16 + m) * 64];
  V16 aLo, aHi;
  aLo.h[0] = *(const v8h*)(xrow + 8 * hl);
  aLo.h[1] = *(const v8h*)(xrow + 16 + 8 * hl);
  aHi.h[0] = *(const v8h*)(xrow + 32 + 8 * hl);
  aHi.h[1] = *(const v8h*)(xrow + 48 + 8 * hl);

#pragma unroll
  for (int nt = 0; nt < 4; ++nt) {
    // B fragment: column n per lane, K contiguous in transposed W
    const _Float16* wrow = &sWt[(nt * 16 + m) * 64];
    V16 bLo, bHi;
    bLo.h[0] = *(const v8h*)(wrow + 8 * hl);
    bLo.h[1] = *(const v8h*)(wrow + 16 + 8 * hl);
    bHi.h[0] = *(const v8h*)(wrow + 32 + 8 * hl);
    bHi.h[1] = *(const v8h*)(wrow + 48 + 8 * hl);

    v8f c = {};
    c = __builtin_amdgcn_wmma_f32_16x16x32_f16(false, aLo.v, false, bLo.v, (short)0, c, false, false);
    c = __builtin_amdgcn_wmma_f32_16x16x32_f16(false, aHi.v, false, bHi.v, (short)0, c, false, false);
#pragma unroll
    for (int r = 0; r < 8; ++r) {
      long rr = row0 + wave * 16 + r + 8 * hl;   // C/D: M = r + 8*(lane/16), N = lane%16
      Hout[rr * kH + nt * 16 + m] = c[r];
    }
  }
}

// ---------------------------------------------------------------- scores
__global__ void k_score(const float* __restrict__ Hm, const float* __restrict__ a,
                        float* __restrict__ sc, int n) {
  int i = blockIdx.x * blockDim.x + threadIdx.x;
  if (i >= n) return;
  const float* h = Hm + (size_t)i * kH;
  float s = 0.f;
#pragma unroll
  for (int j = 0; j < kH; ++j) s += h[j] * a[j];
  sc[i] = s;
}

// ---------------------------------------------------------------- edge phase
__device__ __forceinline__ void atomicMaxF(float* addr, float val) {
  if (val >= 0.f) atomicMax((int*)addr, __float_as_int(val));
  else            atomicMin((unsigned int*)addr, __float_as_uint(val));
}

__global__ void k_edge_logit(const int* __restrict__ src, const int* __restrict__ dst,
                             const float* __restrict__ scs, const float* __restrict__ scd,
                             float* __restrict__ logit, float* __restrict__ mbuf) {
  int e = blockIdx.x * blockDim.x + threadIdx.x;
  int stride = gridDim.x * blockDim.x;
  for (; e < kE; e += stride) {
    float s = scs[src[e]] + scd[dst[e]];
    float lg = s > 0.f ? s : 0.2f * s;      // leaky_relu(.,0.2)
    logit[e] = lg;
    atomicMaxF(&mbuf[dst[e]], lg);
  }
}

__global__ void k_edge_exp(const int* __restrict__ dst, const float* __restrict__ logit,
                           const float* __restrict__ mbuf, float* __restrict__ ex,
                           float* __restrict__ den) {
  int e = blockIdx.x * blockDim.x + threadIdx.x;
  int stride = gridDim.x * blockDim.x;
  for (; e < kE; e += stride) {
    int d = dst[e];
    float v = expf(logit[e] - mbuf[d]);
    ex[e] = v;
    atomicAdd(&den[d], v);
  }
}

// one wave (32 lanes) per edge; 64 features -> 2 atomics per lane
__global__ void k_edge_scatter(const int* __restrict__ src, const int* __restrict__ dst,
                               const float* __restrict__ ex, const float* __restrict__ den,
                               const float* __restrict__ HS, float* __restrict__ OUT) {
  int e = (blockIdx.x * blockDim.x + threadIdx.x) >> 5;
  int lane = threadIdx.x & 31;
  if (e >= kE) return;
  int d = dst[e], s = src[e];
  float alpha = ex[e] / (den[d] + 1e-16f);
  const float* hs = HS + (size_t)s * kH;
  atomicAdd(&OUT[(size_t)d * kH + lane],      alpha * hs[lane]);
  atomicAdd(&OUT[(size_t)d * kH + 32 + lane], alpha * hs[32 + lane]);
}

__global__ void k_bias_act(float* __restrict__ X, long n, const float* __restrict__ b1,
                           const float* __restrict__ b2, int relu) {
  long i = (long)blockIdx.x * blockDim.x + threadIdx.x;
  long stride = (long)gridDim.x * blockDim.x;
  long tot = n * kH;
  for (; i < tot; i += stride) {
    int f = (int)(i & 63);
    float v = X[i] + b1[f] + (b2 ? b2[f] : 0.f);
    if (relu) v = fmaxf(v, 0.f);
    X[i] = v;
  }
}

// ---------------------------------------------------------------- actor head
__global__ void k_ln_logits(const float* __restrict__ P, const float* __restrict__ g,
                            const float* __restrict__ bb, const float* __restrict__ Wo,
                            const float* __restrict__ bo, float* __restrict__ logits) {
  int i = blockIdx.x * blockDim.x + threadIdx.x;
  if (i >= kNP) return;
  const float* p = P + (size_t)i * kH;
  float mu = 0.f;
#pragma unroll
  for (int j = 0; j < kH; ++j) mu += p[j];
  mu *= (1.f / kH);
  float var = 0.f;
#pragma unroll
  for (int j = 0; j < kH; ++j) { float d = p[j] - mu; var += d * d; }
  var *= (1.f / kH);
  float inv = rsqrtf(var + 1e-5f);
  float l0 = bo[0], l1 = bo[1];
#pragma unroll
  for (int j = 0; j < kH; ++j) {
    float ln = (p[j] - mu) * inv * g[j] + bb[j];
    l0 += ln * Wo[j * 2];
    l1 += ln * Wo[j * 2 + 1];
  }
  logits[i * 2]     = l0;
  logits[i * 2 + 1] = l1;
}

// per-batch softmax over 64 contiguous part nodes, 2 channels; writes actions
__global__ void k_actions(const float* __restrict__ logits, float* __restrict__ out) {
  __shared__ float red[64];
  int b = blockIdx.x, id = threadIdx.x;
  float l0 = logits[(b * 64 + id) * 2];
  float l1 = logits[(b * 64 + id) * 2 + 1];

  red[id] = l0; __syncthreads();
  for (int s = 32; s > 0; s >>= 1) { if (id < s) red[id] = fmaxf(red[id], red[id + s]); __syncthreads(); }
  float m0 = red[0]; __syncthreads();
  red[id] = l1; __syncthreads();
  for (int s = 32; s > 0; s >>= 1) { if (id < s) red[id] = fmaxf(red[id], red[id + s]); __syncthreads(); }
  float m1 = red[0]; __syncthreads();

  float e0 = expf(l0 - m0), e1 = expf(l1 - m1);
  red[id] = e0; __syncthreads();
  for (int s = 32; s > 0; s >>= 1) { if (id < s) red[id] += red[id + s]; __syncthreads(); }
  float s0 = red[0]; __syncthreads();
  red[id] = e1; __syncthreads();
  for (int s = 32; s > 0; s >>= 1) { if (id < s) red[id] += red[id + s]; __syncthreads(); }
  float s1 = red[0];

  out[b * 128 + id]      = e0 / (s0 + 1e-16f);
  out[b * 128 + 64 + id] = e1 / (s1 + 1e-16f);
}

// ---------------------------------------------------------------- critic head
__global__ void k_aggr(const float* __restrict__ X, int perB, float* __restrict__ cat, int off) {
  int b = blockIdx.x, f = threadIdx.x;   // 64 threads
  const float* base = X + (size_t)b * perB * kH;
  float mx = -INFINITY, mn = INFINITY, sm = 0.f;
  for (int r = 0; r < perB; ++r) {
    float v = base[(size_t)r * kH + f];
    mx = fmaxf(mx, v); mn = fminf(mn, v); sm += v;
  }
  float* row = cat + (size_t)b * 576 + off;
  row[f] = mx; row[64 + f] = mn; row[128 + f] = sm / perB;
}

__global__ void k_critic1(const float* __restrict__ cat, const float* __restrict__ W,
                          const float* __restrict__ bias, float* __restrict__ c1) {
  int i = blockIdx.x * blockDim.x + threadIdx.x;
  if (i >= kB * kH) return;
  int b = i >> 6, j = i & 63;
  const float* row = cat + (size_t)b * 576;
  float s = bias[j];
  for (int k = 0; k < 576; ++k) s += row[k] * W[k * kH + j];
  c1[i] = 0.5f * s * (1.f + erff(s * 0.70710678118654752f));   // exact gelu
}

__global__ void k_critic2(const float* __restrict__ c1, const float* __restrict__ W,
                          const float* __restrict__ bias, float* __restrict__ c2) {
  int i = blockIdx.x * blockDim.x + threadIdx.x;
  if (i >= kB * kH) return;
  int b = i >> 6, j = i & 63;
  const float* row = c1 + (size_t)b * kH;
  float s = bias[j];
#pragma unroll
  for (int k = 0; k < kH; ++k) s += row[k] * W[k * kH + j];
  c2[i] = s;
}

__global__ void k_critic3(const float* __restrict__ c2, const float* __restrict__ W,
                          const float* __restrict__ bias, float* __restrict__ vout) {
  int b = blockIdx.x * blockDim.x + threadIdx.x;
  if (b >= kB) return;
  const float* row = c2 + (size_t)b * kH;
  float s = bias[0];
#pragma unroll
  for (int k = 0; k < kH; ++k) s += row[k] * W[k];
  vout[b] = tanhf(s);
}

// ---------------------------------------------------------------- launch
extern "C" void kernel_launch(void* const* d_in, const int* in_sizes, int n_in,
                              void* d_out, int out_size, void* d_ws, size_t ws_size,
                              hipStream_t stream) {
  (void)in_sizes; (void)n_in; (void)out_size; (void)ws_size;
  const float* mass        = (const float*)d_in[0];
  const int*   part_x      = (const int*)d_in[1];
  const float* torque_x    = (const float*)d_in[2];
  const float* force_x     = (const float*)d_in[3];
  const int*   ei_pt       = (const int*)d_in[8];
  const int*   ei_tp       = (const int*)d_in[9];
  const int*   ei_pf       = (const int*)d_in[10];
  const int*   ei_fp       = (const int*)d_in[11];
  const float* emb_geom_W  = (const float*)d_in[12];
  const float* emb_state   = (const float*)d_in[13];
  const float* conv_Wsrc   = (const float*)d_in[14];
  const float* conv_Wdst   = (const float*)d_in[15];
  const float* conv_asrc   = (const float*)d_in[16];
  const float* conv_adst   = (const float*)d_in[17];
  const float* conv_b      = (const float*)d_in[18];
  const float* actor_Wsrc  = (const float*)d_in[19];
  const float* actor_Wdst  = (const float*)d_in[20];
  const float* actor_asrc  = (const float*)d_in[21];
  const float* actor_adst  = (const float*)d_in[22];
  const float* actor_b     = (const float*)d_in[23];
  const float* ln_g        = (const float*)d_in[24];
  const float* ln_b        = (const float*)d_in[25];
  const float* actor_out_W = (const float*)d_in[26];
  const float* actor_out_b = (const float*)d_in[27];
  const float* crit_cat_W  = (const float*)d_in[28];
  const float* crit_cat_b  = (const float*)d_in[29];
  const float* crit_mlp_W  = (const float*)d_in[30];
  const float* crit_mlp_b  = (const float*)d_in[31];
  const float* crit_out_W  = (const float*)d_in[32];
  const float* crit_out_b  = (const float*)d_in[33];

  char* wp = (char*)d_ws;
  auto alloc = [&](size_t bytes) -> void* {
    void* p = (void*)wp;
    wp += (bytes + 255) & ~(size_t)255;
    return p;
  };
  float* xpA = (float*)alloc((size_t)kNP * kH * 4);
  float* xpB = (float*)alloc((size_t)kNP * kH * 4);
  float* xtA = (float*)alloc((size_t)kNT * kH * 4);
  float* xtB = (float*)alloc((size_t)kNT * kH * 4);
  float* xfA = (float*)alloc((size_t)kNF * kH * 4);
  float* xfB = (float*)alloc((size_t)kNF * kH * 4);
  _Float16* xp16 = (_Float16*)alloc((size_t)kNP * kH * 2);
  _Float16* xt16 = (_Float16*)alloc((size_t)kNT * kH * 2);
  _Float16* xf16 = (_Float16*)alloc((size_t)kNF * kH * 2);
  float* hs_buf = (float*)alloc((size_t)kNT * kH * 4);
  float* hd_buf = (float*)alloc((size_t)kNT * kH * 4);
  float* sc_s = (float*)alloc((size_t)kNT * 4);
  float* sc_d = (float*)alloc((size_t)kNT * 4);
  float* mbuf = (float*)alloc((size_t)kNT * 4);
  float* den  = (float*)alloc((size_t)kNT * 4);
  float* logit = (float*)alloc((size_t)kE * 4);
  float* exb   = (float*)alloc((size_t)kE * 4);
  _Float16* cws16 = (_Float16*)alloc((size_t)16 * 4096 * 2);   // transposed
  _Float16* cwd16 = (_Float16*)alloc((size_t)16 * 4096 * 2);   // transposed
  _Float16* aws16 = (_Float16*)alloc((size_t)4 * 4096 * 2);    // transposed
  _Float16* awd16 = (_Float16*)alloc((size_t)4 * 4096 * 2);    // transposed
  float* logitsB = (float*)alloc((size_t)kNP * 2 * 4);
  float* catB = (float*)alloc((size_t)kB * 576 * 4);
  float* c1B  = (float*)alloc((size_t)kB * kH * 4);
  float* c2B  = (float*)alloc((size_t)kB * kH * 4);

  // weights -> f16, transposed, once
  k_cvt_wT<<<256, 256, 0, stream>>>(conv_Wsrc, cws16, 16);
  k_cvt_wT<<<256, 256, 0, stream>>>(conv_Wdst, cwd16, 16);
  k_cvt_wT<<<64, 256, 0, stream>>>(actor_Wsrc, aws16, 4);
  k_cvt_wT<<<64, 256, 0, stream>>>(actor_Wdst, awd16, 4);

  // embedding
  k_embed<<<(kNP + 255) / 256, 256, 0, stream>>>(mass, part_x, emb_geom_W, emb_state, xpA);

  const float* curP = xpA; const float* curT = torque_x; const float* curF = force_x;
  float* Pbuf[2] = {xpB, xpA}; int pi = 0;
  float* Tbuf[2] = {xtA, xtB}; int ti = 0;
  float* Fbuf[2] = {xfA, xfB}; int fi = 0;
  const float* critP = nullptr; const float* critT = nullptr; const float* critF = nullptr;
  float* ahP = nullptr;

  for (int l = 0; l < 5; ++l) {
    bool actor = (l == 4);
    float* nP = Pbuf[pi];
    float* nT = actor ? nullptr : Tbuf[ti];
    float* nF = actor ? nullptr : Fbuf[fi];

    // f16 snapshots of current features (GEMM inputs, async-LDS friendly)
    k_cvt_f16<<<2048, 256, 0, stream>>>(curP, xp16, (long)kNP * kH);
    k_cvt_f16<<<2048, 256, 0, stream>>>(curT, xt16, (long)kNT * kH);
    k_cvt_f16<<<2048, 256, 0, stream>>>(curF, xf16, (long)kNF * kH);

    k_fill<<<2048, 256, 0, stream>>>(nP, (long)kNP * kH, 0.f);
    if (!actor) {
      k_fill<<<2048, 256, 0, stream>>>(nT, (long)kNT * kH, 0.f);
      k_fill<<<2048, 256, 0, stream>>>(nF, (long)kNF * kH, 0.f);
    }

    for (int k = 0; k < 4; ++k) {
      if (actor && (k == 0 || k == 2)) continue;  // actor uses only part outputs
      const _Float16* srcX; int ns; const _Float16* dstX; int nd; const int* ei; float* outB;
      switch (k) {
        case 0: srcX = xp16; ns = kNP; dstX = xt16; nd = kNT; ei = ei_pt; outB = nT; break;
        case 1: srcX = xt16; ns = kNT; dstX = xp16; nd = kNP; ei = ei_tp; outB = nP; break;
        case 2: srcX = xp16; ns = kNP; dstX = xf16; nd = kNF; ei = ei_pf; outB = nF; break;
        default: srcX = xf16; ns = kNF; dstX = xp16; nd = kNP; ei = ei_fp; outB = nP; break;
      }
      const _Float16* Ws = actor ? aws16 + (size_t)k * 4096 : cws16 + (size_t)(l * 4 + k) * 4096;
      const _Float16* Wd = actor ? awd16 + (size_t)k * 4096 : cwd16 + (size_t)(l * 4 + k) * 4096;
      const float* a_s = actor ? actor_asrc + k * kH : conv_asrc + (size_t)(l * 4 + k) * kH;
      const float* a_d = actor ? actor_adst + k * kH : conv_adst + (size_t)(l * 4 + k) * kH;
      const int* srcI = ei;
      const int* dstI = ei + kE;

      k_gemm64<<<ns / 128, 256, 0, stream>>>(srcX, Ws, hs_buf, ns);
      k_gemm64<<<nd / 128, 256, 0, stream>>>(dstX, Wd, hd_buf, nd);
      k_score<<<(ns + 255) / 256, 256, 0, stream>>>(hs_buf, a_s, sc_s, ns);
      k_score<<<(nd + 255) / 256, 256, 0, stream>>>(hd_buf, a_d, sc_d, nd);
      k_fill<<<512, 256, 0, stream>>>(mbuf, nd, -INFINITY);
      k_fill<<<512, 256, 0, stream>>>(den, nd, 0.f);
      k_edge_logit<<<2048, 256, 0, stream>>>(srcI, dstI, sc_s, sc_d, logit, mbuf);
      k_edge_exp<<<2048, 256, 0, stream>>>(dstI, logit, mbuf, exb, den);
      k_edge_scatter<<<kE / 8, 256, 0, stream>>>(srcI, dstI, exb, den, hs_buf, outB);
    }

    if (!actor) {
      int relu = (l < 3) ? 1 : 0;
      k_bias_act<<<2048, 256, 0, stream>>>(nT, kNT, conv_b + (size_t)(l * 4 + 0) * kH, nullptr, relu);
      k_bias_act<<<2048, 256, 0, stream>>>(nF, kNF, conv_b + (size_t)(l * 4 + 2) * kH, nullptr, relu);
      k_bias_act<<<2048, 256, 0, stream>>>(nP, kNP, conv_b + (size_t)(l * 4 + 1) * kH,
                                           conv_b + (size_t)(l * 4 + 3) * kH, relu);
      curP = nP; pi ^= 1;
      curT = nT; ti ^= 1;
      curF = nF; fi ^= 1;
      if (l == 3) { critP = curP; critT = curT; critF = curF; }
    } else {
      k_bias_act<<<2048, 256, 0, stream>>>(nP, kNP, actor_b + 1 * kH, actor_b + 3 * kH, 0);
      ahP = nP;
    }
  }

  // actor output
  k_ln_logits<<<(kNP + 255) / 256, 256, 0, stream>>>(ahP, ln_g, ln_b, actor_out_W, actor_out_b, logitsB);
  k_actions<<<kB, 64, 0, stream>>>(logitsB, (float*)d_out);

  // critic
  k_aggr<<<kB, 64, 0, stream>>>(critP, 64, catB, 0);
  k_aggr<<<kB, 64, 0, stream>>>(critT, 256, catB, 192);
  k_aggr<<<kB, 64, 0, stream>>>(critF, 256, catB, 384);
  k_critic1<<<(kB * kH + 255) / 256, 256, 0, stream>>>(catB, crit_cat_W, crit_cat_b, c1B);
  k_critic2<<<(kB * kH + 255) / 256, 256, 0, stream>>>(c1B, crit_mlp_W, crit_mlp_b, c2B);
  k_critic3<<<(kB + 255) / 256, 256, 0, stream>>>(c2B, crit_out_W, crit_out_b, (float*)d_out + kB * 128);
}
